// cross_context_44160853738070
// MI455X (gfx1250) — matrix operand order repaired and checked by reading, not compiled, not run on hardware
//
#include <hip/hip_runtime.h>
#include <hip/hip_bf16.h>
#include <math.h>

typedef __attribute__((ext_vector_type(16))) __bf16 v16bf;
typedef __attribute__((ext_vector_type(8)))  float  v8f;

#define B_SZ   8
#define C_CH   32
#define N_PTS  4096
#define K_NN   16
#define FD     96          // 3*C flattened feature dim (divisible by 32)
#define NEG_INF (-3.4e38f)

// ---------------- workspace layout (bytes) ----------------
#define QX_OFF    ((size_t)0)                      // [B][N][32][3] f32  = 12,582,912
#define YFLAT_OFF ((size_t)12582912)               // [B][N][96]   f32  = 12,582,912
#define YBF_OFF   ((size_t)25165824)               // [B][N][96]   bf16 =  6,291,456
#define SQ_OFF    ((size_t)31457280)               // [B][N]       f32  =    131,072
#define KNN_OFF   ((size_t)31588352)               // [B][N][16]   i32  =  2,097,152

union Frag { v16bf v; unsigned int u[8]; };

// ---------------------------------------------------------------------------
// prep: y [B,C,3,N] -> point-major f32 + bf16 features, squared norms
// ---------------------------------------------------------------------------
__global__ void prep_kernel(const float* __restrict__ y,
                            float* __restrict__ yflat,
                            __bf16* __restrict__ ybf,
                            float* __restrict__ sq) {
    int gid = blockIdx.x * blockDim.x + threadIdx.x;      // b*N + n
    if (gid >= B_SZ * N_PTS) return;
    int b = gid >> 12;                                    // /4096
    int n = gid & (N_PTS - 1);
    const float* yb = y + (size_t)b * FD * N_PTS + n;
    float*  fr = yflat + (size_t)gid * FD;
    __bf16* br = ybf   + (size_t)gid * FD;
    float acc = 0.f;
    #pragma unroll 4
    for (int d = 0; d < FD; ++d) {
        float v = yb[(size_t)d * N_PTS];
        fr[d] = v;
        br[d] = (__bf16)v;
        acc += v * v;
    }
    sq[gid] = acc;
}

// ---------------------------------------------------------------------------
// Qx: vn_linear_leaky_relu(x, Wq, Dq) then channel_equi_vec_normalize.
// One wave32 per point; lane = channel. Cross-lane contractions via ds_bpermute.
// ---------------------------------------------------------------------------
__global__ void qx_kernel(const float* __restrict__ x,
                          const float* __restrict__ Wq,
                          const float* __restrict__ Dq,
                          float* __restrict__ qx) {
    __shared__ float sW[1024], sD[1024];
    int t = threadIdx.x;
    for (int i = t; i < 1024; i += 128) { sW[i] = Wq[i]; sD[i] = Dq[i]; }
    __syncthreads();

    int wave = t >> 5, lane = t & 31;
    int gid = blockIdx.x * 4 + wave;                      // b*N + n
    int b = gid >> 12, n = gid & (N_PTS - 1);

    const float* xb = x + (size_t)b * FD * N_PTS + n;
    float g0 = xb[(size_t)(lane * 3 + 0) * N_PTS];
    float g1 = xb[(size_t)(lane * 3 + 1) * N_PTS];
    float g2 = xb[(size_t)(lane * 3 + 2) * N_PTS];

    float p0 = 0.f, p1 = 0.f, p2 = 0.f;
    for (int d = 0; d < 32; ++d) {
        float w = sW[lane * 32 + d];
        p0 += w * __shfl(g0, d, 32);
        p1 += w * __shfl(g1, d, 32);
        p2 += w * __shfl(g2, d, 32);
    }
    float d0 = 0.f, d1 = 0.f, d2 = 0.f;
    for (int c = 0; c < 32; ++c) {
        float w = sD[lane * 32 + c];
        d0 += w * __shfl(p0, c, 32);
        d1 += w * __shfl(p1, c, 32);
        d2 += w * __shfl(p2, c, 32);
    }
    float dot = p0 * d0 + p1 * d1 + p2 * d2;
    float dsq = d0 * d0 + d1 * d1 + d2 * d2 + 1e-6f;
    float s   = dot / dsq;
    bool pos  = dot >= 0.f;
    float o0 = 0.2f * p0 + 0.8f * (pos ? p0 : (p0 - s * d0));
    float o1 = 0.2f * p1 + 0.8f * (pos ? p1 : (p1 - s * d1));
    float o2 = 0.2f * p2 + 0.8f * (pos ? p2 : (p2 - s * d2));

    float nn = sqrtf(o0 * o0 + o1 * o1 + o2 * o2);
    float n2 = nn * nn;
    #pragma unroll
    for (int m = 1; m < 32; m <<= 1) n2 += __shfl_xor(n2, m, 32);
    float nch = sqrtf(n2);
    float scale = (nn / fmaxf(nn, 1e-12f)) / fmaxf(nch, 1e-12f);

    float* qp = qx + (size_t)gid * FD + lane * 3;
    qp[0] = o0 * scale; qp[1] = o1 * scale; qp[2] = o2 * scale;
}

// ---------------------------------------------------------------------------
// top-16 replace-min insertion (fully unrolled; register resident)
// ---------------------------------------------------------------------------
__device__ __forceinline__ void topk_insert(float (&d)[16], int (&ix)[16],
                                            float& vmin, int& imin,
                                            float cd, int ci) {
    if (cd <= vmin) return;
    #pragma unroll
    for (int t = 0; t < 16; ++t)
        if (t == imin) { d[t] = cd; ix[t] = ci; }
    vmin = d[0]; imin = 0;
    #pragma unroll
    for (int t = 1; t < 16; ++t)
        if (d[t] < vmin) { vmin = d[t]; imin = t; }
}

// ---------------------------------------------------------------------------
// kNN: Gram matrix via v_wmma_f32_16x16x32_bf16 (K=96 -> 3 WMMAs/tile).
// Operands swapped (A = key tile, B = query tile) so the C fragment lands
// candidates for one query per lane in registers (row r+8h, col = lane&15):
// no LDS transpose needed. Ranking key = 2*inner - sq_key (row-constant
// -sq_query dropped). Per-lane top-16, LDS merge per query row.
// 8 waves/block, 1 block per (b, 16-query tile).
// ---------------------------------------------------------------------------
__global__ void knn_kernel(const __bf16* __restrict__ ybf,
                           const float* __restrict__ sq,
                           int* __restrict__ knn) {
    __shared__ float cd[16][256];         // 16 KB : merge candidates (dist)
    __shared__ int   ci[16][256];         // 16 KB : merge candidates (idx)

    int t = threadIdx.x, wave = t >> 5, lane = t & 31;
    int h  = lane >> 4;                   // lane half
    int mn = lane & 15;                   // row (A) / col (B,C) index
    int tilesPerB = N_PTS / 16;
    int b  = blockIdx.x / tilesPerB;
    int i0 = (blockIdx.x % tilesPerB) * 16;
    const __bf16* base = ybf + (size_t)b * N_PTS * FD;

    // ---- B fragments: queries (K=32 x 16 cols each), loop invariant.
    // B k order used here (contiguous, same as A): lane half h -> K = h*16 + 2v (+1), col = lane&15.
    Frag bq[3];
    #pragma unroll
    for (int kk = 0; kk < 3; ++kk) {
        const __bf16* colp = base + (size_t)(i0 + mn) * FD + kk * 32 + h * 16;
        #pragma unroll
        for (int v = 0; v < 8; ++v)
            bq[kk].u[v] = *(const unsigned int*)(colp + 2 * v);
    }

    float best[16]; int bidx[16];
    #pragma unroll
    for (int q = 0; q < 16; ++q) { best[q] = NEG_INF; bidx[q] = 0; }
    float vmin = NEG_INF; int imin = 0;

    for (int jt = wave; jt < tilesPerB; jt += 8) {
        int j0 = jt * 16;
        // prefetch next key tile for this wave (gfx1250 global_prefetch_b8)
        if (jt + 8 < tilesPerB)
            __builtin_prefetch(base + (size_t)(j0 + 128 + mn) * FD, 0, 3);

        v8f c = {};
        #pragma unroll
        for (int kk = 0; kk < 3; ++kk) {
            // A fragment: key rows, lane half h: K pair base = h*16+2v
            Frag ak;
            const __bf16* rowp = base + (size_t)(j0 + mn) * FD + kk * 32;
            #pragma unroll
            for (int v = 0; v < 8; ++v) {
                int kb = (h << 4) + (v << 1);
                ak.u[v] = *(const unsigned int*)(rowp + kb);
            }
            c = __builtin_amdgcn_wmma_f32_16x16x32_bf16(
                    false, ak.v, false, bq[kk].v, (short)0, c, false, false);
        }
        // per-lane: 8 key candidates (rows j0+8h+r) for query i0+mn
        const float4* sp = (const float4*)(sq + b * N_PTS + j0 + 8 * h);
        float4 s0 = sp[0], s1 = sp[1];
        float sqk[8] = { s0.x, s0.y, s0.z, s0.w, s1.x, s1.y, s1.z, s1.w };
        #pragma unroll
        for (int r = 0; r < 8; ++r) {
            float dv = 2.f * c[r] - sqk[r];
            topk_insert(best, bidx, vmin, imin, dv, j0 + 8 * h + r);
        }
    }

    // dump per-lane lists: query row mn gets 16 lists of 16 (2 halves x 8 waves)
    #pragma unroll
    for (int q = 0; q < 16; ++q) {
        int slot = (wave * 2 + h) * 16 + q;
        cd[mn][slot] = best[q];
        ci[mn][slot] = bidx[q];
    }
    __syncthreads();

    if (t < 16) {
        float bd[16]; int bi[16];
        #pragma unroll
        for (int q = 0; q < 16; ++q) { bd[q] = NEG_INF; bi[q] = 0; }
        float vm = NEG_INF; int im = 0;
        for (int s = 0; s < 256; ++s)
            topk_insert(bd, bi, vm, im, cd[t][s], ci[t][s]);
        int* op = knn + ((size_t)b * N_PTS + i0 + t) * K_NN;
        #pragma unroll
        for (int q = 0; q < 16; ++q) op[q] = bi[q];
    }
}

// ---------------------------------------------------------------------------
// fused attention: one wave32 per point, lane = channel. Online softmax.
// Weights interleaved in LDS: {wk_lo, wk_hi, wv_lo, wv_hi} -> 1 ds_load_b128
// per inner step; {dk, dv} -> 1 ds_load_b64.
// ---------------------------------------------------------------------------
__global__ void attn_kernel(const float* __restrict__ x,
                            const float* __restrict__ yflat,
                            const float* __restrict__ qx,
                            const int*   __restrict__ knn,
                            const float* __restrict__ Wk, const float* __restrict__ Dk,
                            const float* __restrict__ Wv, const float* __restrict__ Dv,
                            float* __restrict__ out) {
    __shared__ __align__(16) float sKV[4096];   // [c][d]{wk_lo,wk_hi,wv_lo,wv_hi} 16KB
    __shared__ __align__(16) float sDD[2048];   // [c][c2]{dk,dv}                   8KB
    int t = threadIdx.x;
    for (int i = t; i < 2048; i += 128) {
        int c = i >> 6, d = i & 63;
        int hi = d >> 5, dd = d & 31;
        sKV[c * 128 + dd * 4 + hi]     = Wk[i];
        sKV[c * 128 + dd * 4 + 2 + hi] = Wv[i];
    }
    for (int i = t; i < 1024; i += 128) {
        int c = i >> 5, c2 = i & 31;
        sDD[c * 64 + c2 * 2 + 0] = Dk[i];
        sDD[c * 64 + c2 * 2 + 1] = Dv[i];
    }
    __syncthreads();

    int wave = t >> 5, lane = t & 31;
    int gid = blockIdx.x * 4 + wave;                      // b*N + n
    int b = gid >> 12, n = gid & (N_PTS - 1);

    const float* ctr = yflat + (size_t)gid * FD + lane * 3;
    float c0 = ctr[0], c1 = ctr[1], c2 = ctr[2];
    const float* qp = qx + (size_t)gid * FD + lane * 3;
    float q0 = qp[0], q1 = qp[1], q2 = qp[2];

    const int* nb = knn + (size_t)gid * K_NN;
    // prefetch the 16 gathered neighbor rows (one per lane 0..15)
    if (lane < K_NN) {
        int jj = nb[lane];
        __builtin_prefetch(yflat + ((size_t)b * N_PTS + jj) * FD, 0, 3);
    }

    float m = NEG_INF, l = 0.f, a0 = 0.f, a1 = 0.f, a2 = 0.f;

    for (int kk = 0; kk < K_NN; ++kk) {
        int j = nb[kk];                                   // wave-uniform
        const float* np = yflat + ((size_t)b * N_PTS + j) * FD + lane * 3;
        float g0 = np[0] - c0, g1 = np[1] - c1, g2 = np[2] - c2;

        // p = W @ yg, yg = [nbr-ctr ; ctr] (64 in-channels)
        float pk0 = 0.f, pk1 = 0.f, pk2 = 0.f;
        float pv0 = 0.f, pv1 = 0.f, pv2 = 0.f;
        for (int d = 0; d < 32; ++d) {
            float u0 = __shfl(g0, d, 32), u1 = __shfl(g1, d, 32), u2 = __shfl(g2, d, 32);
            float h0 = __shfl(c0, d, 32), h1 = __shfl(c1, d, 32), h2 = __shfl(c2, d, 32);
            float4 w = *(const float4*)&sKV[lane * 128 + d * 4];
            pk0 += w.x * u0 + w.y * h0;  pk1 += w.x * u1 + w.y * h1;  pk2 += w.x * u2 + w.y * h2;
            pv0 += w.z * u0 + w.w * h0;  pv1 += w.z * u1 + w.w * h1;  pv2 += w.z * u2 + w.w * h2;
        }
        // d = D @ p
        float dk0 = 0.f, dk1 = 0.f, dk2 = 0.f, dv0 = 0.f, dv1 = 0.f, dv2 = 0.f;
        for (int c = 0; c < 32; ++c) {
            float s0 = __shfl(pk0, c, 32), s1 = __shfl(pk1, c, 32), s2 = __shfl(pk2, c, 32);
            float r0 = __shfl(pv0, c, 32), r1 = __shfl(pv1, c, 32), r2 = __shfl(pv2, c, 32);
            float2 wd = *(const float2*)&sDD[lane * 64 + c * 2];
            dk0 += wd.x * s0;  dk1 += wd.x * s1;  dk2 += wd.x * s2;
            dv0 += wd.y * r0;  dv1 += wd.y * r1;  dv2 += wd.y * r2;
        }
        // VN leaky relu (K path)
        float dotk = pk0 * dk0 + pk1 * dk1 + pk2 * dk2;
        float dsqk = dk0 * dk0 + dk1 * dk1 + dk2 * dk2 + 1e-6f;
        float fk = dotk / dsqk; bool posk = dotk >= 0.f;
        float k0f = 0.2f * pk0 + 0.8f * (posk ? pk0 : (pk0 - fk * dk0));
        float k1f = 0.2f * pk1 + 0.8f * (posk ? pk1 : (pk1 - fk * dk1));
        float k2f = 0.2f * pk2 + 0.8f * (posk ? pk2 : (pk2 - fk * dk2));
        // channel-equivariant normalize (cross-lane)
        float nn = sqrtf(k0f * k0f + k1f * k1f + k2f * k2f);
        float n2 = nn * nn;
        #pragma unroll
        for (int msk = 1; msk < 32; msk <<= 1) n2 += __shfl_xor(n2, msk, 32);
        float nch = sqrtf(n2);
        float scl = (nn / fmaxf(nn, 1e-12f)) / fmaxf(nch, 1e-12f);
        // qk dot + per-head (16-lane) reduction
        float qkc = (k0f * q0 + k1f * q1 + k2f * q2) * scl;
        float hs = qkc;
        #pragma unroll
        for (int msk = 1; msk < 16; msk <<= 1) hs += __shfl_xor(hs, msk, 32);
        float s = hs * 0.14433756729740643f;              // 1/sqrt(3*HEAD_C)
        // VN leaky relu (V path)
        float dotv = pv0 * dv0 + pv1 * dv1 + pv2 * dv2;
        float dsqv = dv0 * dv0 + dv1 * dv1 + dv2 * dv2 + 1e-6f;
        float fv = dotv / dsqv; bool posv = dotv >= 0.f;
        float v0f = 0.2f * pv0 + 0.8f * (posv ? pv0 : (pv0 - fv * dv0));
        float v1f = 0.2f * pv1 + 0.8f * (posv ? pv1 : (pv1 - fv * dv1));
        float v2f = 0.2f * pv2 + 0.8f * (posv ? pv2 : (pv2 - fv * dv2));
        // online softmax accumulate
        float mn_ = fmaxf(m, s);
        float sc = __expf(m - mn_);
        float w  = __expf(s - mn_);
        a0 = a0 * sc + w * v0f;
        a1 = a1 * sc + w * v1f;
        a2 = a2 * sc + w * v2f;
        l = l * sc + w;
        m = mn_;
    }
    float inv = 1.f / l;
    const float* xb = x   + (size_t)b * FD * N_PTS + n;
    float*       ob = out + (size_t)b * FD * N_PTS + n;
    ob[(size_t)(lane * 3 + 0) * N_PTS] = xb[(size_t)(lane * 3 + 0) * N_PTS] + a0 * inv;
    ob[(size_t)(lane * 3 + 1) * N_PTS] = xb[(size_t)(lane * 3 + 1) * N_PTS] + a1 * inv;
    ob[(size_t)(lane * 3 + 2) * N_PTS] = xb[(size_t)(lane * 3 + 2) * N_PTS] + a2 * inv;
}

// ---------------------------------------------------------------------------
extern "C" void kernel_launch(void* const* d_in, const int* in_sizes, int n_in,
                              void* d_out, int out_size, void* d_ws, size_t ws_size,
                              hipStream_t stream) {
    const float* x  = (const float*)d_in[0];
    const float* y  = (const float*)d_in[1];
    const float* Wq = (const float*)d_in[2];
    const float* Dq = (const float*)d_in[3];
    const float* Wk = (const float*)d_in[4];
    const float* Dk = (const float*)d_in[5];
    const float* Wv = (const float*)d_in[6];
    const float* Dv = (const float*)d_in[7];
    float* out = (float*)d_out;

    char* ws = (char*)d_ws;
    float*  qx    = (float*)(ws + QX_OFF);
    float*  yflat = (float*)(ws + YFLAT_OFF);
    __bf16* ybf   = (__bf16*)(ws + YBF_OFF);
    float*  sq    = (float*)(ws + SQ_OFF);
    int*    knn   = (int*)(ws + KNN_OFF);

    const int BN = B_SZ * N_PTS;                          // 32768
    prep_kernel<<<BN / 256, 256, 0, stream>>>(y, yflat, ybf, sq);
    qx_kernel  <<<BN / 4, 128, 0, stream>>>(x, Wq, Dq, qx);
    knn_kernel <<<B_SZ * (N_PTS / 16), 256, 0, stream>>>(ybf, sq, knn);
    attn_kernel<<<BN / 4, 128, 0, stream>>>(x, yflat, qx, knn, Wk, Dk, Wv, Dv, out);
}
